// PointNetSetAbstraction_11192684773543
// MI455X (gfx1250) — matrix operand order repaired and checked by reading, not compiled, not run on hardware
//
#include <hip/hip_runtime.h>
#include <stdint.h>

typedef __attribute__((ext_vector_type(16))) __bf16 v16bf;
typedef __attribute__((ext_vector_type(8)))  float  v8f;
typedef __attribute__((ext_vector_type(4)))  unsigned int uint4v;

// Problem constants
constexpr int Bn = 8, Nn = 4096, Dd = 64, Sc = 1024, NS = 32;
constexpr int Mrows = Bn * Sc * NS;        // 262144 GEMM rows
constexpr int K1 = 96;                     // 67 channels padded to 96 (3 k-steps of 32)
constexpr float RAD2 = 0.2f * 0.2f;
constexpr float EPSV = 1e-5f;
constexpr int POS_OUT = Bn * 3 * Sc;       // floats in d_out before feat_out

// Workspace layout (bytes, 256-aligned)
constexpr size_t OFF_CENTERS = 0;                               // 8*1024*3*4   = 98304
constexpr size_t OFF_GI      = 98304;                           // 8*1024*32*4  = 1048576
constexpr size_t OFF_FEATT   = OFF_GI + 1048576;                // 8*4096*64*2  = 4194304
constexpr size_t OFF_W0      = OFF_FEATT + 4194304;             // 64*96*2      = 12288
constexpr size_t OFF_W1      = OFF_W0 + 12288;                  // 64*64*2      = 8192
constexpr size_t OFF_W2      = OFF_W1 + 8192;                   // 128*64*2     = 16384
constexpr size_t OFF_STATS   = OFF_W2 + 16384;                  // 256*4        = 1024
constexpr size_t OFF_X       = OFF_STATS + 1024;                // 262144*96*2  = 50331648
constexpr size_t OFF_Y       = OFF_X + 50331648;                // 262144*128*4 = 134217728

__device__ __forceinline__ unsigned short f2bf(float f) {
    unsigned int u = __float_as_uint(f);
    unsigned int r = u + 0x7fffu + ((u >> 16) & 1u);
    return (unsigned short)(r >> 16);
}

// ---------------------------------------------------------------------------
// 1) Farthest point sampling: 1 block per batch, 256 threads, 16 points/thread
//    in registers; cumulative-distance argmax per step (matches reference).
// ---------------------------------------------------------------------------
__global__ void __launch_bounds__(256)
fps_kernel(const float* __restrict__ pos, float* __restrict__ centers,
           float* __restrict__ pos_out)
{
    const int b = blockIdx.x;
    const int tid = threadIdx.x;
    const float* pb = pos + (size_t)b * 3 * Nn;

    float x[16], y[16], z[16], cum[16];
#pragma unroll
    for (int i = 0; i < 16; ++i) {
        int n = i * 256 + tid;
        x[i] = pb[n];
        y[i] = pb[Nn + n];
        z[i] = pb[2 * Nn + n];
    }

    __shared__ float s_val[8];
    __shared__ int   s_idx[8];
    __shared__ float s_c[3];
    __shared__ int   s_best;

    // first center = point 0
    float cx = pb[0], cy = pb[Nn], cz = pb[2 * Nn];
    unsigned mask = (tid == 0) ? 1u : 0u;
#pragma unroll
    for (int i = 0; i < 16; ++i) {
        float dx = x[i] - cx, dy = y[i] - cy, dz = z[i] - cz;
        cum[i] = dx * dx + dy * dy + dz * dz;
    }
    if (tid == 0) {
        centers[((size_t)b * Sc) * 3 + 0] = cx;
        centers[((size_t)b * Sc) * 3 + 1] = cy;
        centers[((size_t)b * Sc) * 3 + 2] = cz;
        pos_out[((size_t)b * 3 + 0) * Sc] = cx;
        pos_out[((size_t)b * 3 + 1) * Sc] = cy;
        pos_out[((size_t)b * 3 + 2) * Sc] = cz;
    }

    const int wave = tid >> 5, lane = tid & 31;
    for (int st = 1; st < Sc; ++st) {
        // per-thread argmax over unmasked owned points
        float best = -__builtin_inff();
        int bn = 0x7fffffff;
#pragma unroll
        for (int i = 0; i < 16; ++i) {
            if (!((mask >> i) & 1u)) {
                float v = cum[i];
                int n = i * 256 + tid;
                if (v > best || (v == best && n < bn)) { best = v; bn = n; }
            }
        }
        // wave32 argmax
#pragma unroll
        for (int off = 16; off; off >>= 1) {
            float ov = __shfl_xor(best, off, 32);
            int   on = __shfl_xor(bn, off, 32);
            if (ov > best || (ov == best && on < bn)) { best = ov; bn = on; }
        }
        if (lane == 0) { s_val[wave] = best; s_idx[wave] = bn; }
        __syncthreads();
        if (tid < 32) {
            float v = (tid < 8) ? s_val[tid] : -__builtin_inff();
            int   n = (tid < 8) ? s_idx[tid] : 0x7fffffff;
#pragma unroll
            for (int off = 4; off; off >>= 1) {
                float ov = __shfl_xor(v, off, 32);
                int   on = __shfl_xor(n, off, 32);
                if (ov > v || (ov == v && on < n)) { v = ov; n = on; }
            }
            if (tid == 0) s_best = n;
        }
        __syncthreads();
        int bestn = s_best;
        if (tid == (bestn & 255)) {
            int i = bestn >> 8;
            s_c[0] = x[i]; s_c[1] = y[i]; s_c[2] = z[i];
            mask |= 1u << i;
        }
        __syncthreads();
        cx = s_c[0]; cy = s_c[1]; cz = s_c[2];
#pragma unroll
        for (int i = 0; i < 16; ++i) {
            float dx = x[i] - cx, dy = y[i] - cy, dz = z[i] - cz;
            cum[i] += dx * dx + dy * dy + dz * dz;
        }
        if (tid == 0) {
            centers[((size_t)b * Sc + st) * 3 + 0] = cx;
            centers[((size_t)b * Sc + st) * 3 + 1] = cy;
            centers[((size_t)b * Sc + st) * 3 + 2] = cz;
            pos_out[((size_t)b * 3 + 0) * Sc + st] = cx;
            pos_out[((size_t)b * 3 + 1) * Sc + st] = cy;
            pos_out[((size_t)b * 3 + 2) * Sc + st] = cz;
        }
    }
}

// ---------------------------------------------------------------------------
// 2) Transpose features (B,D,N) f32 -> (B,N,D) bf16 so per-point gathers are
//    contiguous 128B rows.
// ---------------------------------------------------------------------------
__global__ void __launch_bounds__(256)
transpose_feat_kernel(const float* __restrict__ feat, unsigned short* __restrict__ featT)
{
    __shared__ float tile[64][65];
    int b = blockIdx.x >> 6;
    int n0 = (blockIdx.x & 63) << 6;
    int tid = threadIdx.x;
    const float* fb = feat + (size_t)b * Dd * Nn;
#pragma unroll
    for (int j = 0; j < 16; ++j) {
        int idx = tid + j * 256;
        int c = idx >> 6, nn = idx & 63;
        tile[c][nn] = fb[(size_t)c * Nn + n0 + nn];
    }
    __syncthreads();
    unsigned* outp = (unsigned*)featT + ((size_t)b * Nn + n0) * 32;
#pragma unroll
    for (int j = 0; j < 8; ++j) {
        int idx = tid + j * 256;
        int nn = idx >> 5, cp = (idx & 31) << 1;
        unsigned lo = f2bf(tile[cp][nn]);
        unsigned hi = f2bf(tile[cp + 1][nn]);
        outp[(size_t)nn * 32 + (idx & 31)] = lo | (hi << 16);
    }
}

// ---------------------------------------------------------------------------
// 3) Ball query: one wave per center. Scan points in index order, take the
//    first 32 within radius (ballot/popc rank packing), pad with the first.
// ---------------------------------------------------------------------------
__global__ void __launch_bounds__(256)
ballquery_kernel(const float* __restrict__ pos, const float* __restrict__ centers,
                 int* __restrict__ gi)
{
    int lane = threadIdx.x & 31;
    int cidx = blockIdx.x * 8 + (threadIdx.x >> 5);
    int b = cidx >> 10;
    const float* pb = pos + (size_t)b * 3 * Nn;
    float cx = centers[(size_t)cidx * 3 + 0];
    float cy = centers[(size_t)cidx * 3 + 1];
    float cz = centers[(size_t)cidx * 3 + 2];
    int* out = gi + (size_t)cidx * NS;

    int cnt = 0, gi0 = 0;
    for (int base = 0; base < Nn && cnt < NS; base += 32) {
        int n = base + lane;
        float dx = pb[n] - cx, dy = pb[Nn + n] - cy, dz = pb[2 * Nn + n] - cz;
        float d2 = dx * dx + dy * dy + dz * dz;
        bool pred = d2 <= RAD2;
        unsigned m = (unsigned)__ballot(pred);
        int rank = cnt + __popc(m & ((1u << lane) - 1u));
        if (pred && rank < NS) out[rank] = n;
        if (cnt == 0 && m != 0) gi0 = base + (__ffs(m) - 1);
        cnt += __popc(m);
    }
    if (cnt < NS && lane >= cnt && lane < NS) out[lane] = gi0;
}

// ---------------------------------------------------------------------------
// 4) Gather into the GEMM activation matrix X[M][96] (bf16 row-major):
//    cols 0..2 = pos - center, 3..66 = feature, 67..95 = zero padding.
//    One wave per row; dword-packed coalesced stores.
// ---------------------------------------------------------------------------
__global__ void __launch_bounds__(256)
gather_kernel(const float* __restrict__ pos, const unsigned short* __restrict__ featT,
              const float* __restrict__ centers, const int* __restrict__ gi,
              unsigned* __restrict__ Xu)
{
    int lane = threadIdx.x & 31;
    int m = blockIdx.x * 8 + (threadIdx.x >> 5);
    int b = m >> 15;
    int rem = m & 32767;
    int cidx = (b << 10) | (rem >> 5);
    int n = gi[m];                                 // gi is flat [cidx][k] == [m]
    const float* pb = pos + (size_t)b * 3 * Nn;
    float pd[3];
    pd[0] = pb[n]          - centers[(size_t)cidx * 3 + 0];
    pd[1] = pb[Nn + n]     - centers[(size_t)cidx * 3 + 1];
    pd[2] = pb[2 * Nn + n] - centers[(size_t)cidx * 3 + 2];
    const unsigned short* fr = featT + ((size_t)b * Nn + n) * 64;
    unsigned* xr = Xu + (size_t)m * (K1 / 2);

    auto colval = [&](int c) -> unsigned {
        if (c < 3)  return (unsigned)f2bf(pd[c]);
        if (c < 67) return (unsigned)fr[c - 3];
        return 0u;
    };
#pragma unroll
    for (int p = 0; p < 2; ++p) {
        int d = lane + p * 32;
        if (d < K1 / 2) {
            unsigned lo = colval(2 * d);
            unsigned hi = colval(2 * d + 1);
            xr[d] = lo | (hi << 16);
        }
    }
}

// ---------------------------------------------------------------------------
// 5) Weight prep: f32 (cout,cin) -> bf16 row-major with cin padded for WMMA.
// ---------------------------------------------------------------------------
__global__ void prep_weights_kernel(const float* __restrict__ w0, const float* __restrict__ w1,
                                    const float* __restrict__ w2,
                                    unsigned short* __restrict__ W0,
                                    unsigned short* __restrict__ W1,
                                    unsigned short* __restrict__ W2)
{
    int tid = threadIdx.x;
    for (int i = tid; i < 64 * 96; i += 256) {
        int o = i / 96, k = i % 96;
        W0[i] = (k < 67) ? f2bf(w0[o * 67 + k]) : (unsigned short)0;
    }
    for (int i = tid; i < 64 * 64; i += 256)  W1[i] = f2bf(w1[i]);
    for (int i = tid; i < 128 * 64; i += 256) W2[i] = f2bf(w2[i]);
}

__global__ void zero_stats_kernel(float* __restrict__ stats) { stats[threadIdx.x] = 0.f; }

// ---------------------------------------------------------------------------
// 6) WMMA GEMM:  Y[M][C] = X[M][K](bf16) * W[C][K]^T(bf16) + bias, f32 accum.
//    8 waves/block split as CW=2 column groups x 4 row slots. Each wave
//    preloads its B (weight) panel into VGPRs ONCE, then streams MT row
//    tiles: per tile only 2*KT b128 A-loads feed KT*NTW WMMAs, removing the
//    per-WMMA load-wait serialization seen in round 1.
// ---------------------------------------------------------------------------
template <int K, int C, int CW, int MT>
__global__ void __launch_bounds__(256)
gemm_wmma_kernel(const unsigned short* __restrict__ X, const unsigned short* __restrict__ W,
                 const float* __restrict__ bias, float* __restrict__ Y)
{
    constexpr int KT  = K / 32;          // k-steps
    constexpr int NTW = C / 16 / CW;     // column tiles per wave
    constexpr int RW  = 8 / CW;          // row slots per block
    int lane = threadIdx.x & 31;
    int wave = threadIdx.x >> 5;
    int colgrp  = wave % CW;
    int rowslot = wave / CW;
    int r  = lane & 15;
    int hi = lane >> 4;

    union F { uint4v u[2]; v16bf v; };

    // Preload resident weight panel
    F bm[KT][NTW];
#pragma unroll
    for (int kt = 0; kt < KT; ++kt) {
#pragma unroll
        for (int t = 0; t < NTW; ++t) {
            int ch = (colgrp * NTW + t) * 16 + r;
            const unsigned short* wrow = W + (size_t)ch * K + kt * 32 + hi * 8;
            bm[kt][t].u[0] = *(const uint4v*)(wrow);
            bm[kt][t].u[1] = *(const uint4v*)(wrow + 16);
        }
    }

    v8f zero = {0.f, 0.f, 0.f, 0.f, 0.f, 0.f, 0.f, 0.f};
    int rowbase = blockIdx.x * (RW * 16 * MT) + rowslot * 16;
#pragma unroll
    for (int mt = 0; mt < MT; ++mt) {
        int row0 = rowbase + mt * (RW * 16);
        const unsigned short* xrow = X + (size_t)(row0 + r) * K;
        F a[KT];
#pragma unroll
        for (int kt = 0; kt < KT; ++kt) {
            a[kt].u[0] = *(const uint4v*)(xrow + kt * 32 + hi * 8);
            a[kt].u[1] = *(const uint4v*)(xrow + kt * 32 + 16 + hi * 8);
        }
        v8f acc[NTW];
#pragma unroll
        for (int t = 0; t < NTW; ++t) acc[t] = zero;
#pragma unroll
        for (int kt = 0; kt < KT; ++kt) {
#pragma unroll
            for (int t = 0; t < NTW; ++t) {
                acc[t] = __builtin_amdgcn_wmma_f32_16x16x32_bf16(
                    false, a[kt].v, false, bm[kt][t].v, (short)0, acc[t], false, false);
            }
        }
#pragma unroll
        for (int t = 0; t < NTW; ++t) {
            int ch = (colgrp * NTW + t) * 16 + r;
            float bia = bias[ch];
#pragma unroll
            for (int q = 0; q < 8; ++q) {
                int row = row0 + q + hi * 8;
                Y[(size_t)row * C + ch] = acc[t][q] + bia;
            }
        }
    }
}

// ---------------------------------------------------------------------------
// 7) Per-channel sum / sum-of-squares over all M rows (BN stats).
// ---------------------------------------------------------------------------
template <int C>
__global__ void __launch_bounds__(256)
reduce_stats_kernel(const float* __restrict__ Y, float* __restrict__ stats)
{
    constexpr int G = 256 / C;
    constexpr int RPB = Mrows / 256;
    int tid = threadIdx.x;
    int c = tid % C, g = tid / C;
    size_t base = (size_t)blockIdx.x * RPB;
    float s = 0.f, s2 = 0.f;
    for (int rr = g; rr < RPB; rr += G) {
        float v = Y[(base + rr) * C + c];
        s += v; s2 += v * v;
    }
    __shared__ float sh[256], sh2[256];
    sh[tid] = s; sh2[tid] = s2;
    __syncthreads();
    if (tid < C) {
#pragma unroll
        for (int gg = 1; gg < G; ++gg) { s += sh[tid + gg * C]; s2 += sh2[tid + gg * C]; }
        atomicAdd(&stats[c], s);
        atomicAdd(&stats[128 + c], s2);
    }
}

// ---------------------------------------------------------------------------
// 8) BN + ReLU + bf16 repack for the next layer's X (C == 64 both times).
// ---------------------------------------------------------------------------
__global__ void __launch_bounds__(256)
normalize_kernel(const float* __restrict__ Y, const float* __restrict__ stats,
                 const float* __restrict__ gam, const float* __restrict__ bet,
                 unsigned* __restrict__ Xu)
{
    int idx = blockIdx.x * 256 + threadIdx.x;
    int m = idx >> 5;
    int cp = (idx & 31) << 1;
    const float inv = 1.0f / (float)Mrows;
    unsigned pk = 0;
#pragma unroll
    for (int j = 0; j < 2; ++j) {
        int c = cp + j;
        float mean = stats[c] * inv;
        float var = stats[128 + c] * inv - mean * mean;
        float scl = gam[c] * rsqrtf(var + EPSV);
        float shf = bet[c] - mean * scl;
        float v = Y[(size_t)m * 64 + c] * scl + shf;
        v = fmaxf(v, 0.f);
        pk |= (unsigned)f2bf(v) << (16 * j);
    }
    Xu[(size_t)m * 32 + (idx & 31)] = pk;
}

// ---------------------------------------------------------------------------
// 9) Final: BN + ReLU + max over the 32 samples -> feat_out (B,128,1024).
// ---------------------------------------------------------------------------
__global__ void __launch_bounds__(128)
final_kernel(const float* __restrict__ Y, const float* __restrict__ stats,
             const float* __restrict__ gam, const float* __restrict__ bet,
             float* __restrict__ out)
{
    int c = threadIdx.x;          // 0..127
    int cidx = blockIdx.x;        // b*1024 + s
    int b = cidx >> 10, s = cidx & 1023;
    const float inv = 1.0f / (float)Mrows;
    float mean = stats[c] * inv;
    float var = stats[128 + c] * inv - mean * mean;
    float scl = gam[c] * rsqrtf(var + EPSV);
    float shf = bet[c] - mean * scl;
    const float* yb = Y + (size_t)cidx * 32 * 128;
    float mx = 0.f;
#pragma unroll
    for (int k = 0; k < 32; ++k) {
        float v = yb[k * 128 + c] * scl + shf;
        mx = fmaxf(mx, fmaxf(v, 0.f));
    }
    out[POS_OUT + ((size_t)(b * 128 + c)) * 1024 + s] = mx;
}

// ---------------------------------------------------------------------------
extern "C" void kernel_launch(void* const* d_in, const int* in_sizes, int n_in,
                              void* d_out, int out_size, void* d_ws, size_t ws_size,
                              hipStream_t stream)
{
    const float* pos  = (const float*)d_in[0];
    const float* feat = (const float*)d_in[1];
    const float* w0  = (const float*)d_in[2];
    const float* b0  = (const float*)d_in[3];
    const float* g0  = (const float*)d_in[4];
    const float* be0 = (const float*)d_in[5];
    const float* w1  = (const float*)d_in[6];
    const float* b1  = (const float*)d_in[7];
    const float* g1  = (const float*)d_in[8];
    const float* be1 = (const float*)d_in[9];
    const float* w2  = (const float*)d_in[10];
    const float* b2  = (const float*)d_in[11];
    const float* g2  = (const float*)d_in[12];
    const float* be2 = (const float*)d_in[13];
    float* out = (float*)d_out;

    char* ws = (char*)d_ws;
    float*          centers = (float*)(ws + OFF_CENTERS);
    int*            gi      = (int*)(ws + OFF_GI);
    unsigned short* featT   = (unsigned short*)(ws + OFF_FEATT);
    unsigned short* W0      = (unsigned short*)(ws + OFF_W0);
    unsigned short* W1      = (unsigned short*)(ws + OFF_W1);
    unsigned short* W2      = (unsigned short*)(ws + OFF_W2);
    float*          stats   = (float*)(ws + OFF_STATS);
    unsigned short* X       = (unsigned short*)(ws + OFF_X);
    float*          Y       = (float*)(ws + OFF_Y);

    fps_kernel<<<Bn, 256, 0, stream>>>(pos, centers, out);
    transpose_feat_kernel<<<Bn * 64, 256, 0, stream>>>(feat, featT);
    ballquery_kernel<<<Bn * Sc / 8, 256, 0, stream>>>(pos, centers, gi);
    prep_weights_kernel<<<1, 256, 0, stream>>>(w0, w1, w2, W0, W1, W2);
    gather_kernel<<<Mrows / 8, 256, 0, stream>>>(pos, featT, centers, gi, (unsigned*)X);

    // Rows per block = (8/CW)*16*MT = 4*16*4 = 256  ->  grid = 1024
    constexpr int GEMM_GRID = Mrows / 256;

    // Layer 1: 67(pad 96) -> 64
    zero_stats_kernel<<<1, 256, 0, stream>>>(stats);
    gemm_wmma_kernel<96, 64, 2, 4><<<GEMM_GRID, 256, 0, stream>>>(X, W0, b0, Y);
    reduce_stats_kernel<64><<<256, 256, 0, stream>>>(Y, stats);
    normalize_kernel<<<Mrows * 32 / 256, 256, 0, stream>>>(Y, stats, g0, be0, (unsigned*)X);

    // Layer 2: 64 -> 64
    zero_stats_kernel<<<1, 256, 0, stream>>>(stats);
    gemm_wmma_kernel<64, 64, 2, 4><<<GEMM_GRID, 256, 0, stream>>>(X, W1, b1, Y);
    reduce_stats_kernel<64><<<256, 256, 0, stream>>>(Y, stats);
    normalize_kernel<<<Mrows * 32 / 256, 256, 0, stream>>>(Y, stats, g1, be1, (unsigned*)X);

    // Layer 3: 64 -> 128, then BN+ReLU+max fused into final kernel
    zero_stats_kernel<<<1, 256, 0, stream>>>(stats);
    gemm_wmma_kernel<64, 128, 2, 4><<<GEMM_GRID, 256, 0, stream>>>(X, W2, b2, Y);
    reduce_stats_kernel<128><<<256, 256, 0, stream>>>(Y, stats);
    final_kernel<<<Bn * Sc, 128, 0, stream>>>(Y, stats, g2, be2, out);
}
